// SRGNN_49898930045082
// MI455X (gfx1250) — matrix operand order repaired and checked by reading, compiled, not run on hardware
//
#include <hip/hip_runtime.h>
#include <hip/hip_bf16.h>

// ---------------------------------------------------------------------------
// SR-GNN forward for MI455X (gfx1250, wave32, WMMA).
// Dominant cost: Z = A[768,128] @ emb.T[128,100000] -> 307MB f32 out
// (13.2us store floor at 23.3 TB/s).  Strategy: split-bf16 (hi/lo) 3-term
// WMMA f32_16x16x32_bf16 GEMM with three independent accumulator chains,
// emb read once from HBM, A tiles software-pipelined through LDS.
// ---------------------------------------------------------------------------

#define HID      128
#define N_ITEMS  100000
#define N_SESS   256
#define N_NODES  5120
#define N_EDGES  10240
#define N_AROWS  768            // 3 * N_SESS stacked projection rows
#define N_CTILES 6250           // 100000 / 16
#define GEMM_WAVES 25           // waves per block; 250 blocks * 25 = 6250

typedef __attribute__((ext_vector_type(16))) __bf16          v16bf;
typedef __attribute__((ext_vector_type(8)))  float           v8f;
typedef __attribute__((ext_vector_type(4)))  float           v4f;
typedef __attribute__((ext_vector_type(8)))  unsigned short  ushort8;

union Frag {
    v16bf v;
    ushort8 h[2];
    unsigned short u[16];
};

__device__ __forceinline__ float sigmoidf(float x) { return 1.0f / (1.0f + __expf(-x)); }

// Round-to-nearest-even split of f32 into bf16 hi + bf16 lo (x ~= hi + lo).
__device__ __forceinline__ void bfsplit(float x, unsigned short& hi, unsigned short& lo) {
    unsigned u = __float_as_uint(x);
    unsigned rh = (u + 0x7FFFu + ((u >> 16) & 1u)) & 0xFFFF0000u;
    hi = (unsigned short)(rh >> 16);
    float r = x - __uint_as_float(rh);
    unsigned ul = __float_as_uint(r);
    lo = (unsigned short)((ul + 0x7FFFu + ((ul >> 16) & 1u)) >> 16);
}

// ---------------------------------------------------------------------------
// (1) Transpose small weight matrices for coalesced streaming in GRU/attn.
//     wT[k*768 + c]   : c<384 -> gru_wih[c][k] ; c>=384 -> gru_whh[c-384][k]
//     w12T[k*256 + d] : d<128 -> w1[d][k]      ; d>=128 -> w2[d-128][k]
// ---------------------------------------------------------------------------
__global__ void k_transpose(const float* __restrict__ wih, const float* __restrict__ whh,
                            const float* __restrict__ w1, const float* __restrict__ w2,
                            float* __restrict__ wT, float* __restrict__ w12T) {
    int t = blockIdx.x * 256 + threadIdx.x;
    if (t < 384 * 128) {
        int row = t / 128, k = t % 128;
        wT[k * 768 + row]       = wih[t];
        wT[k * 768 + 384 + row] = whh[t];
    }
    if (t < 128 * 128) {
        int row = t / 128, k = t % 128;
        w12T[k * 256 + row]       = w1[t];
        w12T[k * 256 + 128 + row] = w2[t];
    }
}

// ---------------------------------------------------------------------------
// (2) h = emb[x]   (block per node, coalesced row copy)
// ---------------------------------------------------------------------------
__global__ void k_gather(const int* __restrict__ x, const float* __restrict__ emb,
                         float* __restrict__ h) {
    int i = blockIdx.x, d = threadIdx.x;
    h[(size_t)i * HID + d] = emb[(size_t)x[i] * HID + d];
}

// ---------------------------------------------------------------------------
// (3) m[dst] += h[src]   (thread = one edge x 4 features, native f32 atomics)
// ---------------------------------------------------------------------------
__global__ void k_scatter(const int* __restrict__ ei, const float* __restrict__ h,
                          float* __restrict__ m) {
    int t = blockIdx.x * 256 + threadIdx.x;           // [0, N_EDGES*32)
    int e = t >> 5;
    int q = (t & 31) * 4;
    int src = ei[e];
    int dst = ei[N_EDGES + e];
    v4f hv = *(const v4f*)(h + (size_t)src * HID + q);
    float* mp = m + (size_t)dst * HID + q;
    unsafeAtomicAdd(mp + 0, hv[0]);
    unsafeAtomicAdd(mp + 1, hv[1]);
    unsafeAtomicAdd(mp + 2, hv[2]);
    unsafeAtomicAdd(mp + 3, hv[3]);
}

// ---------------------------------------------------------------------------
// (4) last_idx[s] = max node index with batch_map == s
// ---------------------------------------------------------------------------
__global__ void k_lastidx(const int* __restrict__ bm, int* __restrict__ last) {
    int i = blockIdx.x * 256 + threadIdx.x;
    if (i < N_NODES) atomicMax(&last[bm[i]], i);
}

// ---------------------------------------------------------------------------
// (5) GRUCell: 8 nodes per block, thread = output dim.  Weights streamed
//     coalesced from transposed layout, activations broadcast from LDS.
// ---------------------------------------------------------------------------
__global__ __launch_bounds__(128) void k_gru(const float* __restrict__ m, const float* __restrict__ h,
                                             const float* __restrict__ wT, float* __restrict__ v) {
    __shared__ float sm[8][HID];
    __shared__ float sh[8][HID];
    const int d = threadIdx.x;
    const int node0 = blockIdx.x * 8;
#pragma unroll
    for (int j = 0; j < 8; ++j) {
        sm[j][d] = m[(size_t)(node0 + j) * HID + d];
        sh[j][d] = h[(size_t)(node0 + j) * HID + d];
    }
    __syncthreads();
    float air[8] = {0.f}, aiz[8] = {0.f}, ain[8] = {0.f};
    float ahr[8] = {0.f}, ahz[8] = {0.f}, ahn[8] = {0.f};
    for (int k = 0; k < HID; ++k) {
        const float* w = wT + k * 768;
        float wir = w[d], wiz = w[128 + d], win = w[256 + d];
        float whr = w[384 + d], whz = w[512 + d], whn = w[640 + d];
#pragma unroll
        for (int j = 0; j < 8; ++j) {
            float mv = sm[j][k], hv = sh[j][k];
            air[j] += mv * wir;  aiz[j] += mv * wiz;  ain[j] += mv * win;
            ahr[j] += hv * whr;  ahz[j] += hv * whz;  ahn[j] += hv * whn;
        }
    }
#pragma unroll
    for (int j = 0; j < 8; ++j) {
        float r = sigmoidf(air[j] + ahr[j]);
        float z = sigmoidf(aiz[j] + ahz[j]);
        float n = tanhf(ain[j] + r * ahn[j]);
        v[(size_t)(node0 + j) * HID + d] = (1.0f - z) * n + z * sh[j][d];
    }
}

// ---------------------------------------------------------------------------
// (6) Attention: alpha_i = sig(vn W1^T + v W2^T + b2) qw^T + qb,
//     s_g[s] += alpha_i * v_i.   8 nodes per block.
// ---------------------------------------------------------------------------
__global__ __launch_bounds__(128) void k_attn(const float* __restrict__ v, const int* __restrict__ bm,
                                              const int* __restrict__ last, const float* __restrict__ w12T,
                                              const float* __restrict__ b2, const float* __restrict__ qw,
                                              const float* __restrict__ qb, float* __restrict__ sg) {
    __shared__ float sv[8][HID];
    __shared__ float svn[8][HID];
    __shared__ float rbuf[8][HID];
    const int d = threadIdx.x;
    const int node0 = blockIdx.x * 8;
#pragma unroll
    for (int j = 0; j < 8; ++j) {
        int i = node0 + j;
        int s = bm[i];
        int li = last[s];
        sv[j][d]  = v[(size_t)i * HID + d];
        svn[j][d] = v[(size_t)li * HID + d];
    }
    __syncthreads();
    float a1[8] = {0.f}, a2[8] = {0.f};
    for (int k = 0; k < HID; ++k) {
        const float* w = w12T + k * 256;
        float w1v = w[d], w2v = w[128 + d];
#pragma unroll
        for (int j = 0; j < 8; ++j) {
            a1[j] += svn[j][k] * w1v;
            a2[j] += sv[j][k] * w2v;
        }
    }
    float b2v = b2[d], qwv = qw[d];
#pragma unroll
    for (int j = 0; j < 8; ++j) rbuf[j][d] = sigmoidf(a1[j] + a2[j] + b2v) * qwv;
    __syncthreads();
    for (int off = 64; off > 0; off >>= 1) {
        if (d < off) {
#pragma unroll
            for (int j = 0; j < 8; ++j) rbuf[j][d] += rbuf[j][d + off];
        }
        __syncthreads();
    }
    float qbv = qb[0];
#pragma unroll
    for (int j = 0; j < 8; ++j) {
        int i = node0 + j;
        int s = bm[i];
        float alpha = rbuf[j][0] + qbv;
        unsafeAtomicAdd(&sg[(size_t)s * HID + d], alpha * sv[j][d]);
    }
}

// ---------------------------------------------------------------------------
// (7) Per session: sh = sig([v_n ; s_g] W3^T); rows of A = sh@{wc,wca,wo}^T,
//     written as split bf16 (hi/lo).  Block per session.
// ---------------------------------------------------------------------------
__global__ __launch_bounds__(128) void k_session(const float* __restrict__ v, const float* __restrict__ sg,
                                                 const int* __restrict__ last, const float* __restrict__ w3,
                                                 const float* __restrict__ wc, const float* __restrict__ wca,
                                                 const float* __restrict__ wo,
                                                 unsigned short* __restrict__ Ahi,
                                                 unsigned short* __restrict__ Alo) {
    __shared__ float cat[2 * HID];
    __shared__ float shv[HID];
    const int s = blockIdx.x, d = threadIdx.x;
    const int li = last[s];
    cat[d]       = v[(size_t)li * HID + d];
    cat[HID + d] = sg[(size_t)s * HID + d];
    __syncthreads();
    float acc = 0.f;
    for (int k = 0; k < 2 * HID; ++k) acc += cat[k] * w3[d * (2 * HID) + k];
    shv[d] = sigmoidf(acc);
    __syncthreads();
    float b1 = 0.f, b2a = 0.f, b3 = 0.f;
    for (int k = 0; k < HID; ++k) {
        float sx = shv[k];
        b1  += sx * wc [d * HID + k];
        b2a += sx * wca[d * HID + k];
        b3  += sx * wo [d * HID + k];
    }
    unsigned short hb, lb;
    bfsplit(b1, hb, lb);
    Ahi[(size_t)(0 * N_SESS + s) * HID + d] = hb;  Alo[(size_t)(0 * N_SESS + s) * HID + d] = lb;
    bfsplit(b2a, hb, lb);
    Ahi[(size_t)(1 * N_SESS + s) * HID + d] = hb;  Alo[(size_t)(1 * N_SESS + s) * HID + d] = lb;
    bfsplit(b3, hb, lb);
    Ahi[(size_t)(2 * N_SESS + s) * HID + d] = hb;  Alo[(size_t)(2 * N_SESS + s) * HID + d] = lb;
}

// ---------------------------------------------------------------------------
// (8) Big GEMM: out[768,100000] = A @ emb.T via v_wmma_f32_16x16x32_bf16,
//     split-bf16 3-term with THREE independent accumulator chains
//     (AhBh / AlBh / AhBl) to break the D->C WMMA dependency chain.
//     One wave = one 16-item column tile; emb loaded once and kept as 4
//     resident K-step B fragments; A row tiles software-pipelined:
//     staging threads hold tile t+1 in VGPRs while tile t is computed,
//     so the global-load latency overlaps a full tile of WMMA work.
//     6250 col tiles = 250 blocks x 25 waves exactly -> EXEC all-ones.
// ---------------------------------------------------------------------------
#define SA_STRIDE 136   // ushort row stride: pads 272B rows -> bank spread

__global__ __launch_bounds__(32 * GEMM_WAVES) void k_gemm(const unsigned short* __restrict__ Ahi,
                                                          const unsigned short* __restrict__ Alo,
                                                          const float* __restrict__ emb,
                                                          float* __restrict__ out) {
    __shared__ unsigned short sAh[16 * SA_STRIDE];
    __shared__ unsigned short sAl[16 * SA_STRIDE];
    const int tid  = threadIdx.x;
    const int wave = tid >> 5;
    const int lane = tid & 31;
    const int n    = lane & 15;
    const int half = lane >> 4;
    const int ct   = blockIdx.x * GEMM_WAVES + wave;   // 0..6249
    const int item = ct * 16 + n;

    // B fragments: lane(n,half), element e -> K = ks*32 + 16*half + e
    // (contiguous K order chosen here, not AMD's table order; A uses the same order)
    // => each lane reads 16 contiguous f32 of emb row `item`, split to bf16.
    Frag Bh[4], Bl[4];
    const float* erow = emb + (size_t)item * HID;
#pragma unroll
    for (int ks = 0; ks < 4; ++ks) {
        const int base = ks * 32 + half * 16;
#pragma unroll
        for (int e = 0; e < 16; e += 4) {
            v4f xv = *(const v4f*)(erow + base + e);
#pragma unroll
            for (int q = 0; q < 4; ++q) bfsplit(xv[q], Bh[ks].u[e + q], Bl[ks].u[e + q]);
        }
    }

    // Software pipeline for the 16x128 hi/lo A tile (8KB): waves 0..7 stage.
    const bool stager = (tid < 256);                   // wave-uniform
    const int  mr = tid >> 4;                          // row 0..15
    const int  cc = (tid & 15) * 8;                    // 8-ushort chunk
    ushort8 ph, pl;
    if (stager) {
        ph = *(const ushort8*)(Ahi + (size_t)mr * HID + cc);
        pl = *(const ushort8*)(Alo + (size_t)mr * HID + cc);
    }

    for (int t = 0; t < 48; ++t) {
        __syncthreads();           // all reads of previous tile complete
        if (stager) {
            *(ushort8*)(sAh + mr * SA_STRIDE + cc) = ph;   // waits on t's load
            *(ushort8*)(sAl + mr * SA_STRIDE + cc) = pl;
            if (t < 47) {          // prefetch t+1; latency hidden by WMMAs
                ph = *(const ushort8*)(Ahi + (size_t)((t + 1) * 16 + mr) * HID + cc);
                pl = *(const ushort8*)(Alo + (size_t)((t + 1) * 16 + mr) * HID + cc);
            }
        }
        __syncthreads();

        v8f acc0 = {0.f, 0.f, 0.f, 0.f, 0.f, 0.f, 0.f, 0.f};
        v8f acc1 = {0.f, 0.f, 0.f, 0.f, 0.f, 0.f, 0.f, 0.f};
        v8f acc2 = {0.f, 0.f, 0.f, 0.f, 0.f, 0.f, 0.f, 0.f};
#pragma unroll
        for (int ks = 0; ks < 4; ++ks) {
            // A fragments: lane row = n; K runs [k0,k0+16) contiguous,
            // k0 = ks*32 + 16*half   (matches B fragment K order)
            Frag Ah, Al;
            const int k0 = ks * 32 + half * 16;
            Ah.h[0] = *(const ushort8*)(sAh + n * SA_STRIDE + k0);
            Ah.h[1] = *(const ushort8*)(sAh + n * SA_STRIDE + k0 + 8);
            Al.h[0] = *(const ushort8*)(sAl + n * SA_STRIDE + k0);
            Al.h[1] = *(const ushort8*)(sAl + n * SA_STRIDE + k0 + 8);
            acc0 = __builtin_amdgcn_wmma_f32_16x16x32_bf16(false, Ah.v, false, Bh[ks].v,
                                                           (short)0, acc0, false, false);
            acc1 = __builtin_amdgcn_wmma_f32_16x16x32_bf16(false, Al.v, false, Bh[ks].v,
                                                           (short)0, acc1, false, false);
            acc2 = __builtin_amdgcn_wmma_f32_16x16x32_bf16(false, Ah.v, false, Bl[ks].v,
                                                           (short)0, acc2, false, false);
        }
        v8f acc = (acc0 + acc1) + acc2;

        // C/D layout: VGPR r -> M = r + 8*half, N = lane%16
#pragma unroll
        for (int r = 0; r < 8; ++r) {
            const int R = t * 16 + r + half * 8;          // stacked row: zi*256 + sess
            out[(size_t)(R >> 8) * (size_t)(N_SESS * N_ITEMS) +
                (size_t)(R & 255) * (size_t)N_ITEMS + item] = acc[r];
        }
    }
}

// ---------------------------------------------------------------------------
// Launch
// ---------------------------------------------------------------------------
extern "C" void kernel_launch(void* const* d_in, const int* in_sizes, int n_in,
                              void* d_out, int out_size, void* d_ws, size_t ws_size,
                              hipStream_t stream) {
    const int*   x     = (const int*)  d_in[0];
    const int*   ei    = (const int*)  d_in[1];
    const int*   bm    = (const int*)  d_in[2];
    const float* emb   = (const float*)d_in[3];
    const float* wih   = (const float*)d_in[4];
    const float* whh   = (const float*)d_in[5];
    const float* w1    = (const float*)d_in[6];
    const float* w2    = (const float*)d_in[7];
    const float* b2    = (const float*)d_in[8];
    const float* qw    = (const float*)d_in[9];
    const float* qb    = (const float*)d_in[10];
    const float* w3    = (const float*)d_in[11];
    const float* wc    = (const float*)d_in[12];
    const float* wca   = (const float*)d_in[13];
    const float* wo    = (const float*)d_in[14];
    float* out = (float*)d_out;

    char* ws = (char*)d_ws;
    float* ws_h    = (float*)(ws + 0);                          // 5120*128 f32
    float* ws_m    = (float*)(ws + 2621440);                    // 5120*128 f32
    float* ws_v    = (float*)(ws + 5242880);                    // 5120*128 f32
    float* ws_wT   = (float*)(ws + 7864320);                    // 128*768 f32
    float* ws_w12T = (float*)(ws + 8257536);                    // 128*256 f32
    int*   ws_last = (int*)  (ws + 8388608);                    // 256 i32
    float* ws_sg   = (float*)(ws + 8389632);                    // 256*128 f32
    unsigned short* ws_Ahi = (unsigned short*)(ws + 8520704);   // 768*128 u16
    unsigned short* ws_Alo = (unsigned short*)(ws + 8717312);   // 768*128 u16

    // zero accumulation targets (graph-capture safe)
    hipMemsetAsync(ws_m,    0, (size_t)N_NODES * HID * sizeof(float), stream);
    hipMemsetAsync(ws_sg,   0, (size_t)N_SESS * HID * sizeof(float), stream);
    hipMemsetAsync(ws_last, 0, (size_t)N_SESS * sizeof(int), stream);

    k_transpose<<<192, 256, 0, stream>>>(wih, whh, w1, w2, ws_wT, ws_w12T);
    k_gather<<<N_NODES, HID, 0, stream>>>(x, emb, ws_h);
    k_scatter<<<(N_EDGES * 32) / 256, 256, 0, stream>>>(ei, ws_h, ws_m);
    k_lastidx<<<(N_NODES + 255) / 256, 256, 0, stream>>>(bm, ws_last);
    k_gru<<<N_NODES / 8, HID, 0, stream>>>(ws_m, ws_h, ws_wT, ws_v);
    k_attn<<<N_NODES / 8, HID, 0, stream>>>(ws_v, bm, ws_last, ws_w12T, b2, qw, qb, ws_sg);
    k_session<<<N_SESS, HID, 0, stream>>>(ws_v, ws_sg, ws_last, w3, wc, wca, wo, ws_Ahi, ws_Alo);
    k_gemm<<<N_CTILES / GEMM_WAVES, 32 * GEMM_WAVES, 0, stream>>>(ws_Ahi, ws_Alo, emb, out);
}
